// ProposalLayer_42417097016361
// MI455X (gfx1250) — matrix operand order repaired and checked by reading, not compiled, not run on hardware
//
#include <hip/hip_runtime.h>
#include <math.h>

typedef __attribute__((ext_vector_type(16))) _Float16 v16h;
typedef __attribute__((ext_vector_type(8)))  float    v8f;

#define B_       2
#define C_IN_    128
#define N_CLS_   2
#define N_YAW_   2
#define DOF_     7
#define NY_      400
#define NX_      352
#define HW_      (NY_*NX_)        /* 140800 */
#define A_       (N_YAW_*HW_)     /* 281600 */
#define TOPK_    256
#define IOU_THR_ 0.01f
#define DEG2RAD_ 0.017453292519943295f

#define NBOX_      (B_*N_CLS_*TOPK_)   /* 1024 */
#define OFF_SCORES (NBOX_*DOF_)        /* 7168 */
#define OFF_BATCH  (OFF_SCORES+NBOX_)  /* 8192 */
#define OFF_CLSI   (OFF_BATCH+NBOX_)   /* 9216 */
#define OFF_KEEP   (OFF_CLSI+NBOX_)    /* 10240 */

// ---------------------------------------------------------------------------
// Kernel 1: classification-head GEMM via WMMA (+bias+sigmoid).
// One wave handles a 16-position spatial strip; M-tile = 16 rows, of which
// rows 0..3 are the cls weights (rest zero). K=128 done as 4x WMMA f16 k=32.
// Memory-bound: reads 144 MB of feature map -> ~6.2us at 23.3 TB/s.
// ---------------------------------------------------------------------------
__global__ void k1_cls_wmma(const float* __restrict__ feat,
                            const float* __restrict__ w_cls,
                            const float* __restrict__ b_cls,
                            float* __restrict__ scores)
{
  const int lane = threadIdx.x & 31;
  const int wave = threadIdx.x >> 5;
  const int tile = blockIdx.x * 8 + wave;
  const int b    = blockIdx.y;
  if (tile >= HW_ / 16) return;
  const int p0   = tile * 16;
  const int ln15 = lane & 15;      // A-row (M) for A-frag; column (N) for B/D
  const int hi   = lane >> 4;

  // A fragments: 16x32 f16 per k-step. Layout (ISA 7.12.2): lane = M row;
  // VGPR v<4 -> K = 2v(+1) (+8 for hi lanes); v>=4 -> K = 16+2(v-4) (+8 hi).
  v16h aw[4];
  #pragma unroll
  for (int ks = 0; ks < 4; ++ks) {
    v16h a = {};
    if (ln15 < 4) {
      #pragma unroll
      for (int v = 0; v < 8; ++v) {
        int kk = ks * 32 + ((v < 4) ? 2 * v : 16 + 2 * (v - 4)) + (hi ? 8 : 0);
        a[2 * v]     = (_Float16)w_cls[ln15 * C_IN_ + kk];
        a[2 * v + 1] = (_Float16)w_cls[ln15 * C_IN_ + kk + 1];
      }
    }
    aw[ks] = a;
  }

  // B fragments: 32x16 f16, same K layout as A: v<4 -> K = 2v(+1) (+8 hi);
  // v>=4 -> K = 16+2(v-4) (+8 hi); lane&15 = column (spatial position).
  const int koff = hi ? 16 : 0;
  const float* fb = feat + (size_t)b * C_IN_ * HW_ + p0 + ln15;
  v8f acc = {};
  #pragma unroll
  for (int ks = 0; ks < 4; ++ks) {
    v16h bf = {};
    #pragma unroll
    for (int v = 0; v < 8; ++v) {
      int c0 = ks * 32 + ((v < 4) ? 2 * v : 16 + 2 * (v - 4)) + (hi ? 8 : 0);
      bf[2 * v]     = (_Float16)fb[(size_t)c0 * HW_];
      bf[2 * v + 1] = (_Float16)fb[(size_t)(c0 + 1) * HW_];
    }
    acc = __builtin_amdgcn_wmma_f32_16x16x32_f16(
        /*neg_a=*/false, aw[ks], /*neg_b=*/false, bf,
        /*c_mod=*/(short)0, acc, /*reuse_a=*/false, /*reuse_b=*/false);
  }

  // D layout: VGPR r, lanes 0-15 -> M=r; lanes 16-31 -> M=r+8 (all zero rows).
  if (!hi) {
    #pragma unroll
    for (int r = 0; r < 4; ++r) {
      float val = acc[r] + b_cls[r];
      float s = 1.0f / (1.0f + __expf(-val));
      int cls = r >> 1, yaw = r & 1;
      scores[(size_t)(b * N_CLS_ + cls) * A_ + (size_t)yaw * HW_ + p0 + ln15] = s;
    }
  }
}

// ---------------------------------------------------------------------------
// Kernel 2: exact top-256 per (b,cls) group via streaming bitonic merge.
// Running current-top-256 (sorted desc) in LDS; each 256-elem chunk is merged
// only if its block-max beats the current 256th score (early skip).
// ---------------------------------------------------------------------------
__global__ void k2_topk(const float* __restrict__ scores,
                        float* __restrict__ tk_score,
                        int*   __restrict__ tk_idx)
{
  __shared__ float skey[2 * TOPK_];
  __shared__ int   sidx[2 * TOPK_];
  __shared__ float red[TOPK_];
  const int tid = threadIdx.x;
  const int g   = blockIdx.x;
  const float* base = scores + (size_t)g * A_;

  skey[tid] = -3.0e38f; skey[tid + TOPK_] = -3.0e38f;
  sidx[tid] = 0;        sidx[tid + TOPK_] = 0;
  __syncthreads();

  for (int c = 0; c < A_ / TOPK_; ++c) {
    float v = base[c * TOPK_ + tid];
    red[tid] = v;
    __syncthreads();
    for (int s = TOPK_ / 2; s > 0; s >>= 1) {
      if (tid < s) red[tid] = fmaxf(red[tid], red[tid + s]);
      __syncthreads();
    }
    float cmax = red[0];
    float cmin = skey[TOPK_ - 1];
    __syncthreads();
    if (cmax <= cmin) continue;     // uniform branch

    skey[TOPK_ + tid] = v;
    sidx[TOPK_ + tid] = c * TOPK_ + tid;
    __syncthreads();

    // bitonic sort of 512, descending
    for (int k = 2; k <= 2 * TOPK_; k <<= 1) {
      for (int j = k >> 1; j > 0; j >>= 1) {
        for (int i = tid; i < 2 * TOPK_; i += TOPK_) {
          int p = i ^ j;
          if (p > i) {
            bool desc = ((i & k) == 0);
            float a = skey[i], bq = skey[p];
            bool sw = desc ? (a < bq) : (a > bq);
            if (sw) {
              skey[i] = bq; skey[p] = a;
              int t = sidx[i]; sidx[i] = sidx[p]; sidx[p] = t;
            }
          }
        }
        __syncthreads();
      }
    }
  }
  tk_score[g * TOPK_ + tid] = skey[tid];
  tk_idx  [g * TOPK_ + tid] = sidx[tid];
}

// ---------------------------------------------------------------------------
// Kernel 3: per-selected-anchor reg GEMV + box decode.
// Recomputing reg only for the 1024 winners avoids writing a 31.5 MB reg map.
// ---------------------------------------------------------------------------
__global__ void k3_decode(const float* __restrict__ feat,
                          const float* __restrict__ anchors,
                          const float* __restrict__ w_reg,
                          const float* __restrict__ b_reg,
                          const float* __restrict__ tk_score,
                          const int*   __restrict__ tk_idx,
                          float* __restrict__ out,
                          float* __restrict__ bev)
{
  const int g = blockIdx.x, k = threadIdx.x;
  const int b = g >> 1, cls = g & 1;
  const int a = tk_idx[g * TOPK_ + k];
  const int yaw = a / HW_;
  const int p   = a - yaw * HW_;
  const int obase = (cls * N_YAW_ + yaw) * DOF_;

  float acc[DOF_];
  #pragma unroll
  for (int d = 0; d < DOF_; ++d) acc[d] = b_reg[obase + d];
  const float* fp = feat + (size_t)b * C_IN_ * HW_ + p;
  #pragma unroll 4
  for (int c = 0; c < C_IN_; ++c) {
    float f = fp[(size_t)c * HW_];
    #pragma unroll
    for (int d = 0; d < DOF_; ++d)
      acc[d] = fmaf(f, w_reg[(obase + d) * C_IN_ + c], acc[d]);
  }

  const float* an = anchors + ((size_t)cls * A_ + a) * DOF_;
  float ax = an[0], ay = an[1], az = an[2];
  float aw = an[3], al = an[4], ah = an[5], ayw = an[6];
  float diag = sqrtf(aw * aw + al * al);

  float bx[DOF_];
  bx[0] = acc[0] * diag + ax;
  bx[1] = acc[1] * diag + ay;
  bx[2] = acc[2] * ah   + az;
  bx[3] = __expf(acc[3]) * aw;
  bx[4] = __expf(acc[4]) * al;
  bx[5] = __expf(acc[5]) * ah;
  bx[6] = acc[6] + ayw;

  const int gi = g * TOPK_ + k;
  #pragma unroll
  for (int d = 0; d < DOF_; ++d) out[gi * DOF_ + d] = bx[d];
  out[OFF_SCORES + gi] = tk_score[gi];
  out[OFF_BATCH  + gi] = (float)b;
  out[OFF_CLSI   + gi] = (float)cls;

  bev[gi * 5 + 0] = bx[0];
  bev[gi * 5 + 1] = bx[1];
  bev[gi * 5 + 2] = bx[3];
  bev[gi * 5 + 3] = bx[4];
  bev[gi * 5 + 4] = bx[6];
}

// ---------------------------------------------------------------------------
// Kernel 4: rotated-BEV IoU greedy NMS (boxes already sorted desc by score).
// ---------------------------------------------------------------------------
__device__ static float rot_inter(const float* b1, const float* b2)
{
  float c1x[4], c1y[4], c2x[4], c2y[4];
  const float dxs[4] = {0.5f, -0.5f, -0.5f, 0.5f};
  const float dys[4] = {0.5f,  0.5f, -0.5f, -0.5f};
  {
    float t = b1[4] * DEG2RAD_, c = cosf(t), s = sinf(t);
    for (int i = 0; i < 4; ++i) {
      float X = dxs[i] * b1[2], Y = dys[i] * b1[3];
      c1x[i] = b1[0] + X * c - Y * s;
      c1y[i] = b1[1] + X * s + Y * c;
    }
  }
  {
    float t = b2[4] * DEG2RAD_, c = cosf(t), s = sinf(t);
    for (int i = 0; i < 4; ++i) {
      float X = dxs[i] * b2[2], Y = dys[i] * b2[3];
      c2x[i] = b2[0] + X * c - Y * s;
      c2y[i] = b2[1] + X * s + Y * c;
    }
  }
  float px[12], py[12];
  int n = 4;
  for (int i = 0; i < 4; ++i) { px[i] = c1x[i]; py[i] = c1y[i]; }

  for (int e = 0; e < 4; ++e) {
    float ax = c2x[e], ay = c2y[e];
    float ex = c2x[(e + 1) & 3] - ax, ey = c2y[(e + 1) & 3] - ay;
    float qx[12], qy[12];
    int m = 0;
    for (int i = 0; i < n; ++i) {
      int j = (i + 1 == n) ? 0 : i + 1;
      float di = ex * (py[i] - ay) - ey * (px[i] - ax);
      float dj = ex * (py[j] - ay) - ey * (px[j] - ax);
      if ((di >= 0.f) != (dj >= 0.f)) {
        float den = di - dj;
        float safe = (fabsf(den) < 1e-12f) ? 1e-12f : den;
        float t = fminf(fmaxf(di / safe, 0.f), 1.f);
        qx[m] = px[i] + t * (px[j] - px[i]);
        qy[m] = py[i] + t * (py[j] - py[i]);
        ++m;
      }
      if (dj >= 0.f) { qx[m] = px[j]; qy[m] = py[j]; ++m; }
    }
    n = m;
    for (int i = 0; i < n; ++i) { px[i] = qx[i]; py[i] = qy[i]; }
    if (n == 0) break;
  }
  if (n < 3) return 0.f;
  float area = 0.f;
  for (int i = 0; i < n; ++i) {
    int j = (i + 1 == n) ? 0 : i + 1;
    area += px[i] * py[j] - py[i] * px[j];
  }
  return 0.5f * fabsf(area);
}

__global__ void k4_nms(const float* __restrict__ bev,
                       const float* __restrict__ tk_score,
                       float* __restrict__ out)
{
  __shared__ float sb[TOPK_][5];
  __shared__ int   skeep[TOPK_];
  const int g = blockIdx.x, tid = threadIdx.x;

  #pragma unroll
  for (int j = 0; j < 5; ++j) sb[tid][j] = bev[(g * TOPK_ + tid) * 5 + j];
  skeep[tid] = 1;
  __syncthreads();

  float myb[5];
  #pragma unroll
  for (int j = 0; j < 5; ++j) myb[j] = sb[tid][j];
  float my_area = myb[2] * myb[3];

  for (int i = 0; i < TOPK_ - 1; ++i) {
    if (skeep[i] && tid > i && skeep[tid]) {
      float bi[5];
      #pragma unroll
      for (int j = 0; j < 5; ++j) bi[j] = sb[i][j];
      float inter = rot_inter(bi, myb);
      float uni = fmaxf(bi[2] * bi[3] + my_area - inter, 1e-12f);
      if (inter / uni > IOU_THR_) skeep[tid] = 0;
    }
    __syncthreads();
  }

  float sc = tk_score[g * TOPK_ + tid];
  int kp = skeep[tid] && (sc > 0.1f);   // SCORE_THRESH = 0.1 for both classes
  out[OFF_KEEP + g * TOPK_ + tid] = kp ? 1.0f : 0.0f;
}

// ---------------------------------------------------------------------------
extern "C" void kernel_launch(void* const* d_in, const int* in_sizes, int n_in,
                              void* d_out, int out_size, void* d_ws, size_t ws_size,
                              hipStream_t stream)
{
  const float* feat    = (const float*)d_in[0];
  const float* anchors = (const float*)d_in[1];
  const float* w_cls   = (const float*)d_in[2];
  const float* b_cls   = (const float*)d_in[3];
  const float* w_reg   = (const float*)d_in[4];
  const float* b_reg   = (const float*)d_in[5];
  float* out = (float*)d_out;

  char* ws = (char*)d_ws;
  float* scores_ws = (float*)ws;
  size_t off = (size_t)B_ * N_CLS_ * A_ * sizeof(float);        // 4.5 MB
  float* tk_score = (float*)(ws + off); off += (size_t)B_ * N_CLS_ * TOPK_ * sizeof(float);
  int*   tk_idx   = (int*)  (ws + off); off += (size_t)B_ * N_CLS_ * TOPK_ * sizeof(int);
  float* bev      = (float*)(ws + off);                         // 1024*5 floats

  dim3 g1((HW_ / 16) / 8, B_);   // 1100 x 2 blocks, 8 waves each
  k1_cls_wmma<<<g1, 256, 0, stream>>>(feat, w_cls, b_cls, scores_ws);
  k2_topk   <<<dim3(B_ * N_CLS_), TOPK_, 0, stream>>>(scores_ws, tk_score, tk_idx);
  k3_decode <<<dim3(B_ * N_CLS_), TOPK_, 0, stream>>>(feat, anchors, w_reg, b_reg,
                                                      tk_score, tk_idx, out, bev);
  k4_nms    <<<dim3(B_ * N_CLS_), TOPK_, 0, stream>>>(bev, tk_score, out);
}
